// MambaPIDTuner_82085414961494
// MI455X (gfx1250) — matrix operand order repaired and checked by reading, not compiled, not run on hardware
//
#include <hip/hip_runtime.h>
#include <hip/hip_bf16.h>
#include <math.h>

typedef __attribute__((ext_vector_type(16))) __bf16 v16bf;
typedef __attribute__((ext_vector_type(8)))  __bf16 v8bf;
typedef __attribute__((ext_vector_type(8)))  float  v8f;

union V16 { v16bf v; v8bf h[2]; };

constexpr int BB = 32, LL = 2048, IN_DIM = 4, D_MODEL = 256, D_INNER = 512;
constexpr int D_STATE = 16, DT_RANK = 16, D_CONV = 4, OUT_DIM = 3;
constexpr int NDBC = DT_RANK + 2 * D_STATE;      // 48
constexpr size_t NROW = (size_t)BB * LL;         // 65536 rows

#ifndef __has_builtin
#define __has_builtin(x) 0
#endif
#if __has_builtin(__builtin_amdgcn_tensor_load_to_lds)
#define HAVE_TDM 1
#else
#define HAVE_TDM 0
#endif

#if HAVE_TDM
typedef __attribute__((ext_vector_type(4))) unsigned tdm_v4u;
typedef __attribute__((ext_vector_type(8))) int      tdm_v8i;
typedef __attribute__((ext_vector_type(4))) int      tdm_v4i;

// Issue one 2-D TDM tile load: `rows` rows of 32 bf16 (64B), row stride
// `stride_elems`, into LDS rows of 40 bf16 (64B data + 16B pad).
// D# per CDNA5 ISA 08_async_tensor §8: group0 = {count=1|type=2, lds, gaddr},
// group1 = {data_size=2B, pad_en, pad_interval=64B, pad_amount=16B, dims}.
__device__ __forceinline__ void tdm_load_tile(unsigned lds_off, const __bf16* gsrc,
                                              unsigned rows, unsigned stride_elems) {
  unsigned long long ga = (unsigned long long)(uintptr_t)gsrc;
  tdm_v4u g0;
  g0[0] = 1u;                                              // count=1, user, no gather
  g0[1] = lds_off;                                         // lds_addr (bytes)
  g0[2] = (unsigned)ga;                                    // global_addr[31:0]
  g0[3] = (unsigned)((ga >> 32) & 0x01ffffffu) | (2u << 30); // addr[56:32] | type=2
  tdm_v8i g1;
  g1[0] = (int)((1u << 16) | (1u << 20) | (3u << 22) | (3u << 25)); // 2B elem, pad 16B/64B
  g1[1] = (int)(32u << 16);                 // tensor_dim0 = 32   (bits 63:48)
  g1[2] = (int)(rows << 16);                // tensor_dim1 = rows (bits 95:80)
  g1[3] = (int)(32u << 16);                 // tile_dim0 = 32     (bits 127:112)
  g1[4] = (int)(rows & 0xffffu);            // tile_dim1 = rows, tile_dim2 = 0
  g1[5] = (int)stride_elems;                // tensor_dim0_stride (elements)
  g1[6] = 0;
  g1[7] = 0;
  tdm_v4i z = {0, 0, 0, 0};
#if __clang_major__ >= 23
  tdm_v8i z8 = {0, 0, 0, 0, 0, 0, 0, 0};
  __builtin_amdgcn_tensor_load_to_lds(g0, g1, z, z, z8, 0);
#else
  __builtin_amdgcn_tensor_load_to_lds(g0, g1, z, z, 0);
#endif
}

__device__ __forceinline__ void tdm_wait0() {
#if __has_builtin(__builtin_amdgcn_s_wait_tensorcnt)
  __builtin_amdgcn_s_wait_tensorcnt(0);
#else
  asm volatile("s_wait_tensorcnt 0x0" ::: "memory");
#endif
}
#endif  // HAVE_TDM

// ---------------------------------------------------------------------------
// Weight prep: bf16 + transpose to N-major so B tiles are row-major 2D tiles.
//   wb1t[n][k] = W_inproj[k][n]   (n<512: xc half only), 512x256
//   wb2t[n][k] = W_xproj[k][n],   48x512
// ---------------------------------------------------------------------------
__global__ void k_cvt_weights(const float* __restrict__ w1, const float* __restrict__ w2,
                              __bf16* __restrict__ o1t, __bf16* __restrict__ o2t) {
  size_t i = (size_t)blockIdx.x * blockDim.x + threadIdx.x;
  const size_t n1 = (size_t)D_INNER * D_MODEL;   // 131072
  const size_t n2 = (size_t)NDBC * D_INNER;      // 24576
  if (i < n1) {
    int n = (int)(i / D_MODEL), k = (int)(i % D_MODEL);
    o1t[i] = (__bf16)w1[(size_t)k * (2 * D_INNER) + n];
  } else if (i < n1 + n2) {
    size_t j = i - n1;
    int n = (int)(j / D_INNER), k = (int)(j % D_INNER);
    o2t[j] = (__bf16)w2[(size_t)k * NDBC + n];
  }
}

// ---------------------------------------------------------------------------
// h = x @ W_in + b_in   (K=4, VALU, bf16 out for the WMMA GEMM)
// ---------------------------------------------------------------------------
__global__ void k_in_embed(const float* __restrict__ x, const float* __restrict__ W_in,
                           const float* __restrict__ b_in, __bf16* __restrict__ h) {
  size_t idx = (size_t)blockIdx.x * 256 + threadIdx.x;   // NROW*256 threads
  size_t row = idx >> 8;
  int m = (int)(idx & 255);
  const float* xr = x + row * IN_DIM;
  float acc = b_in[m];
#pragma unroll
  for (int k = 0; k < IN_DIM; ++k) acc += xr[k] * W_in[k * D_MODEL + m];
  h[idx] = (__bf16)acc;
}

// ---------------------------------------------------------------------------
// silu(z) at t = L-1 only, full fp32 (recomputes h_last from x directly)
// ---------------------------------------------------------------------------
__global__ void k_zlast_gate(const float* __restrict__ x, const float* __restrict__ W_in,
                             const float* __restrict__ b_in, const float* __restrict__ W_inproj,
                             float* __restrict__ zsilu) {
  __shared__ float hl[D_MODEL];
  int b = blockIdx.x;
  int tid = threadIdx.x;                     // 512 threads
  size_t row = (size_t)b * LL + (LL - 1);
  if (tid < D_MODEL) {
    float acc = b_in[tid];
#pragma unroll
    for (int k = 0; k < IN_DIM; ++k) acc += x[row * IN_DIM + k] * W_in[k * D_MODEL + tid];
    hl[tid] = acc;
  }
  __syncthreads();
  float acc = 0.f;
  for (int k = 0; k < D_MODEL; ++k)
    acc += hl[k] * W_inproj[(size_t)k * (2 * D_INNER) + D_INNER + tid];
  zsilu[b * D_INNER + tid] = acc / (1.f + __expf(-acc));
}

// ---------------------------------------------------------------------------
// WMMA GEMM: C(fp32) = A(bf16, M x K) x Bt(bf16, N x K pre-transposed)
// block tile 128 x BN, 8 waves (WROWS x WCOLS), K-step 32.
// Tiles staged via TDM tensor_load_to_lds (double-buffered, software
// pipelined, TENSORcnt-tracked) with a plain-copy fallback.
// ---------------------------------------------------------------------------
template <int BN, int WROWS, int WCOLS>
__global__ __launch_bounds__(256)
void k_gemm_bf16(const __bf16* __restrict__ A, const __bf16* __restrict__ Bg,
                 float* __restrict__ C, int lda, int ldb, int ldc, int K) {
  constexpr int BM  = 128;
  constexpr int WTM = BM / WROWS;
  constexpr int WTN = BN / WCOLS;
  constexpr int MT  = WTM / 16;
  constexpr int NT  = WTN / 16;
  __shared__ __align__(16) __bf16 As[2][BM][40];   // 128 x 32 (+16B row pad)
  __shared__ __align__(16) __bf16 Bt[2][BN][40];   // [n][k]  (+16B row pad)

  const int tid  = threadIdx.x;
  const int wid  = tid >> 5;
  const int lane = tid & 31;
  const int half = lane >> 4;
  const int l16  = lane & 15;
  const int waveM = (wid / WCOLS) * WTM;
  const int waveN = (wid % WCOLS) * WTN;
  const size_t m0 = (size_t)blockIdx.y * BM;
  const int n0 = blockIdx.x * BN;

  v8f acc[MT][NT];
  for (int i = 0; i < MT; ++i)
    for (int j = 0; j < NT; ++j)
      for (int r = 0; r < 8; ++r) acc[i][j][r] = 0.f;

  auto stage = [&](int k0, int buf) {
#if HAVE_TDM
    if (tid < 32) {   // wave 0 issues both DMA tiles (EXEC-independent, per-wave)
      tdm_load_tile((unsigned)(uintptr_t)&As[buf][0][0], A + m0 * lda + k0, BM, (unsigned)lda);
      tdm_load_tile((unsigned)(uintptr_t)&Bt[buf][0][0], Bg + (size_t)n0 * ldb + k0, BN, (unsigned)ldb);
    }
#else
    for (int i = tid; i < BM * 4; i += 256) {
      int row = i >> 2, c = (i & 3) * 8;
      *(v8bf*)&As[buf][row][c] = *(const v8bf*)(A + (m0 + row) * lda + k0 + c);
    }
    for (int i = tid; i < BN * 4; i += 256) {
      int row = i >> 2, c = (i & 3) * 8;
      *(v8bf*)&Bt[buf][row][c] = *(const v8bf*)(Bg + ((size_t)n0 + row) * ldb + k0 + c);
    }
#endif
  };

  const int NIT = K / 32;
  stage(0, 0);
  for (int it = 0; it < NIT; ++it) {
    const int cur = it & 1;
#if HAVE_TDM
    if (tid < 32) tdm_wait0();     // tiles for this iteration have landed
#endif
    __syncthreads();               // ... and are visible to all 8 waves
    if (it + 1 < NIT) stage((it + 1) * 32, cur ^ 1);  // overlap next DMA with math

    // A frag (ISA 16-bit A 16x32 layout): lane half -> K 0..7/8..15, 16..23/24..31
    V16 a[MT], b[NT];
#pragma unroll
    for (int fm = 0; fm < MT; ++fm) {
      const __bf16* r = &As[cur][waveM + fm * 16 + l16][0];
      a[fm].h[0] = *(const v8bf*)(r + half * 8);
      a[fm].h[1] = *(const v8bf*)(r + 16 + half * 8);
    }
    // B frag (same K order as A): lane half -> K 0..7/8..15, 16..23/24..31, col = l16
#pragma unroll
    for (int fn = 0; fn < NT; ++fn) {
      const __bf16* r = &Bt[cur][waveN + fn * 16 + l16][0];
      b[fn].h[0] = *(const v8bf*)(r + half * 8);
      b[fn].h[1] = *(const v8bf*)(r + 16 + half * 8);
    }
#pragma unroll
    for (int fm = 0; fm < MT; ++fm)
#pragma unroll
      for (int fn = 0; fn < NT; ++fn)
        acc[fm][fn] = __builtin_amdgcn_wmma_f32_16x16x32_bf16(
            false, a[fm].v, false, b[fn].v, (short)0, acc[fm][fn], false, false);
  }

  // D layout: VGPR r -> row half*8 + r, column = l16
  float* cp = C + (m0 + waveM + half * 8) * (size_t)ldc + n0 + waveN + l16;
#pragma unroll
  for (int fm = 0; fm < MT; ++fm)
#pragma unroll
    for (int fn = 0; fn < NT; ++fn)
#pragma unroll
      for (int r = 0; r < 8; ++r)
        cp[(size_t)(fm * 16 + r) * ldc + fn * 16] = acc[fm][fn][r];
}

// ---------------------------------------------------------------------------
// causal depthwise conv (k=4) + SiLU; fp32 for the scan, bf16 for GEMM-2
// ---------------------------------------------------------------------------
__global__ void k_conv_silu(const float* __restrict__ xc, const float* __restrict__ cw,
                            const float* __restrict__ cb, float* __restrict__ xcv,
                            __bf16* __restrict__ xcv_bf) {
  size_t idx = (size_t)blockIdx.x * 256 + threadIdx.x;   // NROW*512
  int d = (int)(idx & (D_INNER - 1));
  size_t row = idx >> 9;                                 // b*L + l
  int l = (int)(row & (LL - 1));
  float acc = cb[d];
#pragma unroll
  for (int k = 0; k < D_CONV; ++k) {
    int t = l - (D_CONV - 1) + k;
    if (t >= 0) acc += cw[d * D_CONV + k] * xc[(row + (size_t)(t - l)) * D_INNER + d];
  }
  float v = acc / (1.f + __expf(-acc));
  xcv[idx] = v;
  xcv_bf[idx] = (__bf16)v;
}

// ---------------------------------------------------------------------------
// dt = softplus(dtr @ W_dt + b_dt), K=16, fp32
// ---------------------------------------------------------------------------
__global__ void k_dt(const float* __restrict__ dbc, const float* __restrict__ W_dt,
                     const float* __restrict__ b_dt, float* __restrict__ dt) {
  size_t idx = (size_t)blockIdx.x * 256 + threadIdx.x;   // NROW*512
  int d = (int)(idx & (D_INNER - 1));
  size_t row = idx >> 9;
  const float* dr = dbc + row * NDBC;
  float acc = b_dt[d];
#pragma unroll
  for (int r = 0; r < DT_RANK; ++r) acc += dr[r] * W_dt[r * D_INNER + d];
  dt[idx] = (acc > 20.f) ? acc : log1pf(__expf(acc));
}

// ---------------------------------------------------------------------------
// selective scan: thread per (b, d, n); only the last-step y survives (per
// the reference's out[:, -1]), so C-contraction/gate happen once at t=L-1.
// ---------------------------------------------------------------------------
__global__ __launch_bounds__(1024)
void k_scan(const float* __restrict__ dt, const float* __restrict__ xcv,
            const float* __restrict__ dbc, const float* __restrict__ A_log,
            const float* __restrict__ Dvec, const float* __restrict__ zsilu,
            float* __restrict__ y) {
  int b  = blockIdx.y;
  int n  = threadIdx.x & (D_STATE - 1);
  int dl = threadIdx.x >> 4;                 // 0..63
  int d  = blockIdx.x * 64 + dl;
  float a = -__expf(A_log[d * D_STATE + n]);
  float h = 0.f;
  size_t base = (size_t)b * LL;
  const float* pdt = dt  + base * D_INNER + d;
  const float* pxv = xcv + base * D_INNER + d;
  const float* pB  = dbc + base * NDBC + DT_RANK + n;
  for (int t = 0; t < LL; ++t) {
    if (t + 8 < LL) {
      __builtin_prefetch(pdt + 8 * D_INNER, 0, 0);
      __builtin_prefetch(pxv + 8 * D_INNER, 0, 0);
    }
    float dtv = *pdt;
    float xv  = *pxv;
    float Bv  = *pB;
    h = __expf(dtv * a) * h + (dtv * xv) * Bv;
    pdt += D_INNER; pxv += D_INNER; pB += NDBC;
  }
  size_t lrow = base + (LL - 1);
  float part = h * dbc[lrow * NDBC + DT_RANK + D_STATE + n];   // C at last step
  part += __shfl_xor(part, 8, 16);
  part += __shfl_xor(part, 4, 16);
  part += __shfl_xor(part, 2, 16);
  part += __shfl_xor(part, 1, 16);
  if (n == 0) {
    float yv = part + xcv[lrow * D_INNER + d] * Dvec[d];
    y[b * D_INNER + d] = yv * zsilu[b * D_INNER + d];
  }
}

// ---------------------------------------------------------------------------
// out = (y @ W_outproj) @ W_out + b_out, last timestep only
// ---------------------------------------------------------------------------
__global__ void k_head(const float* __restrict__ y, const float* __restrict__ Wo,
                       const float* __restrict__ W_out, const float* __restrict__ b_out,
                       float* __restrict__ out) {
  __shared__ float om[D_MODEL];
  int b = blockIdx.x, m = threadIdx.x;       // 256 threads
  float acc = 0.f;
  for (int d = 0; d < D_INNER; ++d) acc += y[b * D_INNER + d] * Wo[(size_t)d * D_MODEL + m];
  om[m] = acc;
  __syncthreads();
  if (m < OUT_DIM) {
    float o = b_out[m];
    for (int k = 0; k < D_MODEL; ++k) o += om[k] * W_out[k * OUT_DIM + m];
    out[b * OUT_DIM + m] = o;
  }
}

// ---------------------------------------------------------------------------
extern "C" void kernel_launch(void* const* d_in, const int* in_sizes, int n_in,
                              void* d_out, int out_size, void* d_ws, size_t ws_size,
                              hipStream_t stream) {
  const float* x      = (const float*)d_in[0];
  const float* W_in   = (const float*)d_in[1];
  const float* b_in   = (const float*)d_in[2];
  const float* W_inpj = (const float*)d_in[3];
  const float* conv_w = (const float*)d_in[4];
  const float* conv_b = (const float*)d_in[5];
  const float* W_xpj  = (const float*)d_in[6];
  const float* W_dt   = (const float*)d_in[7];
  const float* b_dt   = (const float*)d_in[8];
  const float* A_log  = (const float*)d_in[9];
  const float* Dv     = (const float*)d_in[10];
  const float* W_opj  = (const float*)d_in[11];
  const float* W_out  = (const float*)d_in[12];
  const float* b_out  = (const float*)d_in[13];
  float* out = (float*)d_out;
  char*  ws  = (char*)d_ws;

  const size_t sz_hbf   = NROW * D_MODEL * sizeof(__bf16);  // 32 MB
  const size_t sz_xc    = NROW * D_INNER * sizeof(float);   // 128 MB
  const size_t sz_xcv   = NROW * D_INNER * sizeof(float);   // 128 MB
  const size_t sz_xcvbf = NROW * D_INNER * sizeof(__bf16);  // 64 MB
  const size_t sz_dbc   = NROW * NDBC * sizeof(float);      // 12.6 MB

  __bf16* h_buf    = (__bf16*)(ws + 0);
  float*  dt_buf   = (float*)(ws + 0);               // aliases h_bf + xc (dead by k_dt)
  float*  xc_buf   = (float*)(ws + sz_hbf);
  float*  xcv_buf  = (float*)(ws + sz_hbf + sz_xc);
  __bf16* xcvb_buf = (__bf16*)(ws + sz_hbf + sz_xc + sz_xcv);
  float*  dbc_buf  = (float*)(ws + sz_hbf + sz_xc + sz_xcv + sz_xcvbf);
  char*   p        = ws + sz_hbf + sz_xc + sz_xcv + sz_xcvbf + sz_dbc;
  __bf16* wb1t     = (__bf16*)p;  p += (size_t)D_INNER * D_MODEL * 2;
  __bf16* wb2t     = (__bf16*)p;  p += (size_t)NDBC * D_INNER * 2;
  float*  zsilu    = (float*)p;   p += (size_t)BB * D_INNER * 4;
  float*  y_buf    = (float*)p;

  k_cvt_weights<<<(131072 + 24576 + 255) / 256, 256, 0, stream>>>(W_inpj, W_xpj, wb1t, wb2t);
  k_in_embed<<<(unsigned)(NROW * D_MODEL / 256), 256, 0, stream>>>(x, W_in, b_in, h_buf);
  k_zlast_gate<<<BB, D_INNER, 0, stream>>>(x, W_in, b_in, W_inpj, zsilu);
  // xc = h @ W_inproj[:, :512]   (M=65536, N=512, K=256)
  k_gemm_bf16<128, 4, 2><<<dim3(D_INNER / 128, (unsigned)(NROW / 128)), 256, 0, stream>>>(
      h_buf, wb1t, xc_buf, D_MODEL, D_MODEL, D_INNER, D_MODEL);
  k_conv_silu<<<(unsigned)(NROW * D_INNER / 256), 256, 0, stream>>>(xc_buf, conv_w, conv_b,
                                                                   xcv_buf, xcvb_buf);
  // dbc = xcv @ W_xproj          (M=65536, N=48, K=512)
  k_gemm_bf16<48, 8, 1><<<dim3(1, (unsigned)(NROW / 128)), 256, 0, stream>>>(
      xcvb_buf, wb2t, dbc_buf, D_INNER, D_INNER, NDBC, D_INNER);
  k_dt<<<(unsigned)(NROW * D_INNER / 256), 256, 0, stream>>>(dbc_buf, W_dt, b_dt, dt_buf);
  k_scan<<<dim3(D_INNER / 64, BB), 1024, 0, stream>>>(dt_buf, xcv_buf, dbc_buf, A_log, Dv, zsilu, y_buf);
  k_head<<<BB, D_MODEL, 0, stream>>>(y_buf, W_opj, W_out, b_out, out);
}
